// Encoder_76459007803482
// MI455X (gfx1250) — matrix operand order repaired and checked by reading, not compiled, not run on hardware
//
#include <hip/hip_runtime.h>

typedef _Float16 v16h __attribute__((ext_vector_type(16)));
typedef _Float16 v8h  __attribute__((ext_vector_type(8)));
typedef float    v8f  __attribute__((ext_vector_type(8)));
typedef int      v4i  __attribute__((ext_vector_type(4)));

#define A_   12
#define H_   6
#define D_   192
#define T_   80
#define B_   8
#define CC_  192
#define NSEQ 96
#define G_   640
#define E_   132
#define E2_  144
#define NT_  7680     // NSEQ*T_ == G_*A_
#define HC_  1152     // H_*CC_

#ifndef __has_builtin
#define __has_builtin(x) 0
#endif
#if __has_builtin(__builtin_amdgcn_global_load_async_to_lds_b128) && \
    __has_builtin(__builtin_amdgcn_s_wait_asynccnt)
#define ASYNC_LDS 1
typedef __attribute__((address_space(1))) v4i* as1_v4i_ptr;
typedef __attribute__((address_space(3))) v4i* as3_v4i_ptr;
#else
#define ASYNC_LDS 0
#endif

static inline int cdiv_h(int a, int b) { return (a + b - 1) / b; }

// ---------------------------------------------------------------------------
// Block-cooperative WMMA GEMM: 8 waves per block compute a 128x64 patch.
// A: f16 row-major. Bsw: f16 fragment-swizzled (see convw_kernel).
// The 4-ntile B slab (4KB) is staged into LDS per K-step, double-buffered,
// via async-to-LDS when the toolchain exposes it (ASYNCcnt-tracked).
// ---------------------------------------------------------------------------
template <int ACT, bool RES, typename OutT>
__global__ __launch_bounds__(256) void gemm_wmma(const _Float16* __restrict__ A,
                                                 const _Float16* __restrict__ Bsw,
                                                 const float* __restrict__ bias,
                                                 const float* __restrict__ res,
                                                 OutT* __restrict__ Cout,
                                                 int M, int N, int K) {
  const int tid  = threadIdx.x;
  const int lane = tid & 31;
  const int w    = tid >> 5;                 // wave 0..7
  const int l15  = lane & 15, half = lane >> 4;
  const int ntiles = N >> 4;
  const int nblk   = N >> 6;                 // 64-col blocks
  const int bm  = blockIdx.x / nblk;
  const int bn  = blockIdx.x % nblk;
  const int nt0 = bn << 2;
  const int mt  = (bm << 3) + w;
  const int m   = (mt << 4) + l15;
  const int ksteps = K >> 5;

  __shared__ __align__(32) _Float16 Bs[2][4 * 32 * 16];   // 8 KB, double-buffered

  // each thread stages 8 halfs (16 bytes): slab layout [nt2][lane][16]
  const int s_nt   = tid >> 6;               // 0..3
  const int s_lane = (tid >> 1) & 31;
  const int s_e    = (tid & 1) << 3;
  const int lds_off = (s_nt * 32 + s_lane) * 16 + s_e;

  auto stage = [&](int kk, int buf) {
    const _Float16* src =
        Bsw + (((size_t)kk * ntiles + nt0 + s_nt) * 32 + s_lane) * 16 + s_e;
    _Float16* dst = &Bs[buf][lds_off];
#if ASYNC_LDS
    __builtin_amdgcn_global_load_async_to_lds_b128(
        (as1_v4i_ptr)(uintptr_t)src, (as3_v4i_ptr)(uintptr_t)dst, 0, 0);
#else
    *(v8h*)dst = *(const v8h*)src;
#endif
  };

  v8f acc[4] = {};
  const _Float16* arow = A + (size_t)m * K + half * 16;

  stage(0, 0);
#if ASYNC_LDS
  __builtin_amdgcn_s_wait_asynccnt(0);
#endif
  __syncthreads();

  for (int kk = 0; kk < ksteps; ++kk) {
    if (kk + 1 < ksteps) stage(kk + 1, (kk + 1) & 1);
    union { v16h v; v8h h[2]; } af;                   // A layout: lane -> row l&15
    af.h[0] = *(const v8h*)(arow + kk * 32);          // k = 32kk+16h .. +7
    af.h[1] = *(const v8h*)(arow + kk * 32 + 8);      // k = 32kk+16h+8 .. +15
    const _Float16* bbase = &Bs[kk & 1][lane * 16];
    for (int t = 0; t < 4; ++t) {
      v16h bf = *(const v16h*)(bbase + t * 512);
      acc[t] = __builtin_amdgcn_wmma_f32_16x16x32_f16(false, af.v, false, bf,
                                                      (short)0, acc[t], false, false);
    }
#if ASYNC_LDS
    __builtin_amdgcn_s_wait_asynccnt(0);
#endif
    __syncthreads();
  }

  for (int t = 0; t < 4; ++t) {
    const int n = ((nt0 + t) << 4) + l15;
    const float bv = bias ? bias[n] : 0.f;
    for (int r = 0; r < 8; ++r) {
      int mr = (mt << 4) + r + (half << 3);
      float v = acc[t][r] + bv;
      if constexpr (ACT == 1) v = 0.5f * v * (1.f + erff(v * 0.70710678118f));
      if constexpr (RES) v += res[(size_t)mr * N + n];
      Cout[(size_t)mr * N + n] = (OutT)v;
    }
  }
}

// Convert f32 weight [K,N] -> f16 fragment-swizzled:
// sw[((kk*ntiles+nt)*32+lane)*16 + e] = W[kk*32 + e + 16*(lane>>4)][nt*16 + (lane&15)]
__global__ void convw_kernel(const float* __restrict__ W, _Float16* __restrict__ sw,
                             int K, int N) {
  int idx = blockIdx.x * blockDim.x + threadIdx.x;
  if (idx >= K * N) return;
  int e    = idx & 15;
  int lane = (idx >> 4) & 31;
  int rest = idx >> 9;
  int ntiles = N >> 4;
  int nt = rest % ntiles;
  int kk = rest / ntiles;
  int k    = kk * 32 + e + 16 * (lane >> 4);
  int ncol = nt * 16 + (lane & 15);
  sw[idx] = (_Float16)W[(size_t)k * N + ncol];
}

// ---------------------------------------------------------------------------
// LayerNorm over D_=192, writes f16 (optionally remapped to GAT node order)
// ---------------------------------------------------------------------------
__global__ __launch_bounds__(192) void ln_kernel(const float* __restrict__ x,
                                                 const float* __restrict__ g,
                                                 const float* __restrict__ b,
                                                 _Float16* __restrict__ out,
                                                 int node_order) {
  int row = blockIdx.x, tid = threadIdx.x;
  float v = x[(size_t)row * D_ + tid];
  float s = v, s2 = v * v;
  for (int o = 16; o > 0; o >>= 1) { s += __shfl_xor(s, o); s2 += __shfl_xor(s2, o); }
  __shared__ float red[12];
  if ((tid & 31) == 0) { red[tid >> 5] = s; red[6 + (tid >> 5)] = s2; }
  __syncthreads();
  float S = 0.f, S2 = 0.f;
  for (int i = 0; i < 6; ++i) { S += red[i]; S2 += red[6 + i]; }
  float mean = S * (1.f / 192.f);
  float var  = S2 * (1.f / 192.f) - mean * mean;
  float xn = (v - mean) * rsqrtf(var + 1e-5f) * g[tid] + b[tid];
  size_t orow = row;
  if (node_order) {
    int nseq = row / T_, t = row % T_;
    int bb = nseq / A_, a = nseq % A_;
    orow = (size_t)(bb * T_ + t) * A_ + a;
  }
  out[orow * D_ + tid] = (_Float16)xn;
}

// ---------------------------------------------------------------------------
// Causal MHA with WMMA. One block per (seq, head); 5 waves, one query tile each.
// ---------------------------------------------------------------------------
__global__ __launch_bounds__(160) void attn_kernel(const float* __restrict__ qkv,
                                                   _Float16* __restrict__ o) {
  const int n  = blockIdx.x / H_;
  const int h  = blockIdx.x % H_;
  const int wv = threadIdx.x >> 5;      // query tile 0..4
  const int lane = threadIdx.x & 31;
  const int l15 = lane & 15, half = lane >> 4;
  __shared__ float    S[5][16][80];
  __shared__ _Float16 P[5][16][96];
  const float scale = 0.1767766953f;    // 1/sqrt(32)
  const float* base = qkv + (size_t)n * T_ * 576;

  // Q fragment (A layout): lane l -> row m=l&15; k = e+16*half
  v16h aq;
  {
    const float* qrow = base + (size_t)(wv * 16 + l15) * 576 + h * 32;
    for (int e = 0; e < 16; ++e)
      aq[e] = (_Float16)qrow[e + 16 * half];
  }
  // scores for key tiles kj <= wv
  for (int kj = 0; kj <= wv; ++kj) {
    int tk = kj * 16 + l15;
    const float* krow = base + (size_t)tk * 576 + D_ + h * 32;
    v16h bf;                      // B layout: lane l -> col n=l&15; k = e+16*half
    for (int e = 0; e < 16; ++e) bf[e] = (_Float16)krow[e + 16 * half];
    v8f c = {};
    c = __builtin_amdgcn_wmma_f32_16x16x32_f16(false, aq, false, bf,
                                               (short)0, c, false, false);
    for (int r = 0; r < 8; ++r) {
      int ml = r + half * 8;
      int tq = wv * 16 + ml;
      S[wv][ml][tk] = (tk <= tq) ? c[r] * scale : -1e30f;
    }
  }
  __syncthreads();
  // row softmax (lanes 0..15 own one row each)
  if (lane < 16) {
    int ml = lane, tq = wv * 16 + ml;
    float mx = -1e30f;
    for (int j = 0; j <= tq; ++j) mx = fmaxf(mx, S[wv][ml][j]);
    float den = 0.f;
    for (int j = 0; j <= tq; ++j) den += expf(S[wv][ml][j] - mx);
    float inv = 1.f / den;
    for (int j = 0; j < 96; ++j) {
      float p = (j <= tq) ? expf(S[wv][ml][j] - mx) * inv : 0.f;
      P[wv][ml][j] = (_Float16)p;
    }
  }
  __syncthreads();
  // O = P[16,96] @ V[96,32] : two 16x16 output tiles, 3 K-steps
  for (int dt = 0; dt < 2; ++dt) {
    v8f c = {};
    for (int ks = 0; ks < 3; ++ks) {
      v16h af, bf;
      for (int e = 0; e < 16; ++e) {
        int key = ks * 32 + e + 16 * half;
        af[e] = P[wv][l15][key];
      }
      int dh = dt * 16 + l15;
      for (int e = 0; e < 16; ++e) {
        int key = ks * 32 + e + 16 * half;
        bf[e] = (key < T_) ? (_Float16)base[(size_t)key * 576 + 2 * D_ + h * 32 + dh]
                           : (_Float16)0.f;
      }
      c = __builtin_amdgcn_wmma_f32_16x16x32_f16(false, af, false, bf,
                                                 (short)0, c, false, false);
    }
    for (int r = 0; r < 8; ++r) {
      int tq = wv * 16 + r + half * 8;
      o[((size_t)n * T_ + tq) * D_ + h * 32 + dt * 16 + l15] = (_Float16)c[r];
    }
  }
}

// ---------------------------------------------------------------------------
// Embedding MLP stages (scalar; tiny FLOPs)
// ---------------------------------------------------------------------------
__global__ void embed1_kernel(const float* __restrict__ sf, const int* __restrict__ aid,
                              const float* __restrict__ emb,
                              const float* __restrict__ W, const float* __restrict__ bL,
                              const float* __restrict__ g, const float* __restrict__ bb,
                              const float* __restrict__ mm, const float* __restrict__ vv,
                              float* __restrict__ out) {
  int idx = blockIdx.x * blockDim.x + threadIdx.x;
  if (idx >= NT_ * 64) return;
  int row = idx >> 6, c = idx & 63;
  int nseq = row / T_;
  int id = aid[nseq]; if (id < 0) id = 0;
  float acc = bL[c];
  for (int k = 0; k < 4; ++k)  acc += sf[(size_t)row * 4 + k] * W[k * 64 + c];
  for (int k = 0; k < 12; ++k) acc += emb[id * 12 + k] * W[(4 + k) * 64 + c];
  float bn = (acc - mm[c]) * rsqrtf(vv[c] + 1e-5f) * g[c] + bb[c];
  out[idx] = fmaxf(bn, 0.f);
}

__global__ void embed2_kernel(const float* __restrict__ in,
                              const float* __restrict__ W, const float* __restrict__ bL,
                              const float* __restrict__ g, const float* __restrict__ bb,
                              const float* __restrict__ mm, const float* __restrict__ vv,
                              float* __restrict__ out) {
  int idx = blockIdx.x * blockDim.x + threadIdx.x;
  if (idx >= NT_ * 256) return;
  int row = idx >> 8, c = idx & 255;
  float acc = bL[c];
  for (int k = 0; k < 64; ++k) acc += in[(size_t)row * 64 + k] * W[k * 256 + c];
  float bn = (acc - mm[c]) * rsqrtf(vv[c] + 1e-5f) * g[c] + bb[c];
  out[idx] = fmaxf(bn, 0.f);
}

__global__ void embed3_kernel(const float* __restrict__ in,
                              const float* __restrict__ W, const float* __restrict__ bL,
                              const float* __restrict__ g, const float* __restrict__ bb,
                              const float* __restrict__ mm, const float* __restrict__ vv,
                              float* __restrict__ xinit, float* __restrict__ x) {
  int idx = blockIdx.x * blockDim.x + threadIdx.x;
  if (idx >= NT_ * D_) return;
  int row = idx / D_, c = idx % D_;
  int t = row % T_;
  float acc = bL[c];
  for (int k = 0; k < 256; ++k) acc += in[(size_t)row * 256 + k] * W[k * D_ + c];
  float bn = (acc - mm[c]) * rsqrtf(vv[c] + 1e-5f) * g[c] + bb[c];
  float val = fmaxf(bn, 0.f);
  xinit[idx] = val;
  float df = expf(-logf(10000.f) * (float)(c & ~1) / (float)D_);
  float pe = (c & 1) ? cosf((float)t * df) : sinf((float)t * df);
  x[idx] = val + pe;
}

// ---------------------------------------------------------------------------
// GAT static structure: src/dst with self loops, per-node incoming lists
// ---------------------------------------------------------------------------
__global__ void meta_kernel(const int* __restrict__ edge_index,
                            int* __restrict__ src2, int* __restrict__ dst2,
                            int* __restrict__ inc) {
  if (blockIdx.x || threadIdx.x) return;
  for (int e = 0; e < E_; ++e) { src2[e] = edge_index[e]; dst2[e] = edge_index[E_ + e]; }
  for (int a = 0; a < A_; ++a) { src2[E_ + a] = a; dst2[E_ + a] = a; }
  for (int a = 0; a < A_; ++a) {
    int cnt = 0;
    for (int e = 0; e < E2_ && cnt < 12; ++e)
      if (dst2[e] == a) inc[a * 12 + cnt++] = e;
  }
}

__global__ void loopea_kernel(const float* __restrict__ ea, const int* __restrict__ inc,
                              float* __restrict__ loopea) {
  int idx = blockIdx.x * blockDim.x + threadIdx.x;
  if (idx >= G_ * A_) return;
  int g = idx / A_, a = idx % A_;
  float s0 = 0.f, s1 = 0.f; int cnt = 0;
  for (int k = 0; k < 12; ++k) {
    int e = inc[a * 12 + k];
    if (e < E_) {
      s0 += ea[((size_t)g * E_ + e) * 2];
      s1 += ea[((size_t)g * E_ + e) * 2 + 1];
      ++cnt;
    }
  }
  loopea[idx * 2]     = s0 / (float)cnt;
  loopea[idx * 2 + 1] = s1 / (float)cnt;
}

// per-(edge, head) attention logit; one wave each
__global__ void alpha_kernel(const _Float16* __restrict__ XL, const _Float16* __restrict__ XR,
                             const float* __restrict__ gwe, const float* __restrict__ gatt,
                             const float* __restrict__ ea, const float* __restrict__ loopea,
                             const int* __restrict__ src2, const int* __restrict__ dst2,
                             float* __restrict__ alpha) {
  int wid  = (blockIdx.x * blockDim.x + threadIdx.x) >> 5;
  int lane = threadIdx.x & 31;
  if (wid >= G_ * E2_ * H_) return;
  int h = wid % H_, e2 = wid / H_;
  int g = e2 / E2_, e = e2 % E2_;
  float a0, a1;
  if (e < E_) {
    a0 = ea[((size_t)g * E_ + e) * 2];
    a1 = ea[((size_t)g * E_ + e) * 2 + 1];
  } else {
    int a = e - E_;
    a0 = loopea[(g * A_ + a) * 2];
    a1 = loopea[(g * A_ + a) * 2 + 1];
  }
  int srow = g * A_ + src2[e], drow = g * A_ + dst2[e];
  float acc = 0.f;
  for (int c = lane; c < CC_; c += 32) {
    float msg = (float)XL[(size_t)srow * HC_ + h * CC_ + c]
              + (float)XR[(size_t)drow * HC_ + h * CC_ + c]
              + a0 * gwe[h * CC_ + c] + a1 * gwe[HC_ + h * CC_ + c];
    msg = (msg > 0.f) ? msg : 0.2f * msg;
    acc += msg * gatt[h * CC_ + c];
  }
  for (int o = 16; o > 0; o >>= 1) acc += __shfl_xor(acc, o);
  if (lane == 0) alpha[(size_t)e2 * H_ + h] = acc;
}

// per-node segment softmax + aggregation + head mean + residual
__global__ __launch_bounds__(192) void agg_kernel(const _Float16* __restrict__ XL,
                                                  const float* __restrict__ alpha,
                                                  const float* __restrict__ gbias,
                                                  const int* __restrict__ src2,
                                                  const int* __restrict__ inc,
                                                  float* __restrict__ x) {
  int node = blockIdx.x;
  int g = node / A_, a = node % A_;
  int tid = threadIdx.x;
  __shared__ int   le_s[12], sr_s[12];
  __shared__ float w_s[6][12];
  if (tid < 12) {
    int e = inc[a * 12 + tid];
    le_s[tid] = e;
    sr_s[tid] = g * A_ + src2[e];
  }
  __syncthreads();
  if (tid < 6) {
    float al[12], mx = -1e30f;
    for (int k = 0; k < 12; ++k) {
      al[k] = alpha[(size_t)(g * E2_ + le_s[k]) * H_ + tid];
      mx = fmaxf(mx, al[k]);
    }
    float den = 0.f;
    for (int k = 0; k < 12; ++k) { al[k] = expf(al[k] - mx); den += al[k]; }
    float inv = 1.f / den;
    for (int k = 0; k < 12; ++k) w_s[tid][k] = al[k] * inv;
  }
  __syncthreads();
  int c = tid;
  float acc = 0.f;
  for (int hh = 0; hh < H_; ++hh)
    for (int k = 0; k < 12; ++k)
      acc += w_s[hh][k] * (float)XL[(size_t)sr_s[k] * HC_ + hh * CC_ + c];
  float val = acc * (1.f / (float)H_) + gbias[c];
  int t = g % T_, bb = g / T_;
  size_t tok = (size_t)(bb * A_ + a) * T_ + t;
  x[tok * D_ + c] += val;
}

// ---------------------------------------------------------------------------
extern "C" void kernel_launch(void* const* d_in, const int* in_sizes, int n_in,
                              void* d_out, int out_size, void* d_ws, size_t ws_size,
                              hipStream_t stream) {
  (void)in_sizes; (void)n_in; (void)out_size; (void)ws_size;
  const float* state_feat = (const float*)d_in[0];
  const int*   agent_ids  = (const int*)d_in[2];
  const int*   edge_index = (const int*)d_in[3];
  const float* edge_attr  = (const float*)d_in[4];
  const float* emb_table  = (const float*)d_in[5];
  const float* laW1 = (const float*)d_in[6],  *lab1 = (const float*)d_in[7];
  const float* bn1g = (const float*)d_in[8],  *bn1b = (const float*)d_in[9];
  const float* bn1m = (const float*)d_in[10], *bn1v = (const float*)d_in[11];
  const float* laW2 = (const float*)d_in[12], *lab2 = (const float*)d_in[13];
  const float* bn2g = (const float*)d_in[14], *bn2b = (const float*)d_in[15];
  const float* bn2m = (const float*)d_in[16], *bn2v = (const float*)d_in[17];
  const float* laW3 = (const float*)d_in[18], *lab3 = (const float*)d_in[19];
  const float* bn3g = (const float*)d_in[20], *bn3b = (const float*)d_in[21];
  const float* bn3m = (const float*)d_in[22], *bn3v = (const float*)d_in[23];
  const float* ln1g = (const float*)d_in[24], *ln1b = (const float*)d_in[25];
  const float* qkvw = (const float*)d_in[26], *qkvb = (const float*)d_in[27];
  const float* outw = (const float*)d_in[28], *outb = (const float*)d_in[29];
  const float* ln2g = (const float*)d_in[30], *ln2b = (const float*)d_in[31];
  const float* fw1  = (const float*)d_in[32], *fb1  = (const float*)d_in[33];
  const float* fw2  = (const float*)d_in[34], *fb2  = (const float*)d_in[35];
  const float* gwl  = (const float*)d_in[36], *gbl  = (const float*)d_in[37];
  const float* gwr  = (const float*)d_in[38], *gbr  = (const float*)d_in[39];
  const float* gwe  = (const float*)d_in[40];
  const float* gatt = (const float*)d_in[41];
  const float* gbias = (const float*)d_in[42];
  const float* ngp  = (const float*)d_in[43], *nbp = (const float*)d_in[44];
  float* out = (float*)d_out;

  // workspace bump allocator (256B aligned)
  char* wp = (char*)d_ws;
  auto alloc = [&](size_t bytes) -> char* {
    char* p = wp; wp += (bytes + 255) & ~(size_t)255; return p;
  };
  float*     X      = (float*)alloc((size_t)NT_ * D_ * 4);
  _Float16*  XN     = (_Float16*)alloc((size_t)NT_ * D_ * 2);
  float*     QKV    = (float*)alloc((size_t)NT_ * 576 * 4);
  _Float16*  OF16   = (_Float16*)alloc((size_t)NT_ * D_ * 2);
  _Float16*  H1     = (_Float16*)alloc((size_t)NT_ * 768 * 2);
  _Float16*  WF16   = (_Float16*)alloc((size_t)192 * HC_ * 2);
  _Float16*  XL     = (_Float16*)alloc((size_t)NT_ * HC_ * 2);
  _Float16*  XR     = (_Float16*)alloc((size_t)NT_ * HC_ * 2);
  float*     ALPHA  = (float*)alloc((size_t)G_ * E2_ * H_ * 4);
  float*     LOOPEA = (float*)alloc((size_t)G_ * A_ * 2 * 4);
  int*       SRC2   = (int*)alloc(E2_ * 4);
  int*       DST2   = (int*)alloc(E2_ * 4);
  int*       INC    = (int*)alloc(A_ * 12 * 4);

  float* H64  = ALPHA;   // reuse (free until GAT): 2.2MB >= 1.97MB
  float* H256 = QKV;     // reuse: 17.7MB >= 7.9MB

  // ---- static graph structure ----
  meta_kernel<<<1, 32, 0, stream>>>(edge_index, SRC2, DST2, INC);
  loopea_kernel<<<cdiv_h(G_ * A_, 256), 256, 0, stream>>>(edge_attr, INC, LOOPEA);

  // ---- embedding MLP -> x_initial (d_out) and x = x_initial + PE ----
  embed1_kernel<<<cdiv_h(NT_ * 64, 256), 256, 0, stream>>>(
      state_feat, agent_ids, emb_table, laW1, lab1, bn1g, bn1b, bn1m, bn1v, H64);
  embed2_kernel<<<cdiv_h(NT_ * 256, 256), 256, 0, stream>>>(
      H64, laW2, lab2, bn2g, bn2b, bn2m, bn2v, H256);
  embed3_kernel<<<cdiv_h(NT_ * D_, 256), 256, 0, stream>>>(
      H256, laW3, lab3, bn3g, bn3b, bn3m, bn3v, out, X);

  auto gemm_blocks = [](int M, int N) { return (M >> 7) * (N >> 6); };

  for (int l = 0; l < 3; ++l) {
    // ---- causal temporal self-attention ----
    ln_kernel<<<NT_, 192, 0, stream>>>(X, ln1g + l * D_, ln1b + l * D_, XN, 0);
    convw_kernel<<<cdiv_h(192 * 576, 256), 256, 0, stream>>>(qkvw + (size_t)l * 192 * 576, WF16, 192, 576);
    gemm_wmma<0, false, float><<<gemm_blocks(NT_, 576), 256, 0, stream>>>(
        XN, WF16, qkvb + l * 576, nullptr, QKV, NT_, 576, 192);
    attn_kernel<<<NSEQ * H_, 160, 0, stream>>>(QKV, OF16);
    convw_kernel<<<cdiv_h(192 * 192, 256), 256, 0, stream>>>(outw + (size_t)l * 192 * 192, WF16, 192, 192);
    gemm_wmma<0, true, float><<<gemm_blocks(NT_, 192), 256, 0, stream>>>(
        OF16, WF16, outb + l * D_, X, X, NT_, 192, 192);

    // ---- FFN (GELU) ----
    ln_kernel<<<NT_, 192, 0, stream>>>(X, ln2g + l * D_, ln2b + l * D_, XN, 0);
    convw_kernel<<<cdiv_h(192 * 768, 256), 256, 0, stream>>>(fw1 + (size_t)l * 192 * 768, WF16, 192, 768);
    gemm_wmma<1, false, _Float16><<<gemm_blocks(NT_, 768), 256, 0, stream>>>(
        XN, WF16, fb1 + l * 768, nullptr, H1, NT_, 768, 192);
    convw_kernel<<<cdiv_h(768 * 192, 256), 256, 0, stream>>>(fw2 + (size_t)l * 768 * 192, WF16, 768, 192);
    gemm_wmma<0, true, float><<<gemm_blocks(NT_, 192), 256, 0, stream>>>(
        H1, WF16, fb2 + l * D_, X, X, NT_, 192, 768);

    // ---- framewise GATv2 ----
    ln_kernel<<<NT_, 192, 0, stream>>>(X, ngp + l * D_, nbp + l * D_, XN, 1);
    convw_kernel<<<cdiv_h(192 * HC_, 256), 256, 0, stream>>>(gwl + (size_t)l * 192 * HC_, WF16, 192, HC_);
    gemm_wmma<0, false, _Float16><<<gemm_blocks(NT_, HC_), 256, 0, stream>>>(
        XN, WF16, gbl + l * HC_, nullptr, XL, NT_, HC_, 192);
    convw_kernel<<<cdiv_h(192 * HC_, 256), 256, 0, stream>>>(gwr + (size_t)l * 192 * HC_, WF16, 192, HC_);
    gemm_wmma<0, false, _Float16><<<gemm_blocks(NT_, HC_), 256, 0, stream>>>(
        XN, WF16, gbr + l * HC_, nullptr, XR, NT_, HC_, 192);
    alpha_kernel<<<cdiv_h(G_ * E2_ * H_, 8), 256, 0, stream>>>(
        XL, XR, gwe + (size_t)l * 2 * HC_, gatt + (size_t)l * HC_,
        edge_attr, LOOPEA, SRC2, DST2, ALPHA);
    agg_kernel<<<NT_, 192, 0, stream>>>(XL, ALPHA, gbias + l * D_, SRC2, INC, X);
  }

  // second output: final x
  (void)hipMemcpyAsync(out + (size_t)NT_ * D_, X, (size_t)NT_ * D_ * 4,
                       hipMemcpyDeviceToDevice, stream);
}
